// PytorchMultiHeadedAttention_7499012899674
// MI455X (gfx1250) — compile-verified, matrix operand order checked by reading, not run on hardware
//
#include <hip/hip_runtime.h>

// ---------------------------------------------------------------------------
// MI455X (gfx1250) multi-head attention.
//   B=4, L=1024, NM=1024, NH=16, ND=64. scale = 1/sqrt(NM) = 1/32.
// bf16 WMMA everywhere; GEMMs are LDS-staged and double-buffered through the
// Tensor Data Mover (tensor_load_to_lds + s_wait_tensorcnt); attention keeps
// the full 32x1024 f32 score stripe in the 320KB LDS.
// ---------------------------------------------------------------------------

typedef __attribute__((ext_vector_type(16))) __bf16 v16bf;
typedef __attribute__((ext_vector_type(8)))  float  v8f;
typedef __attribute__((ext_vector_type(4)))  unsigned int u32x4;
typedef __attribute__((ext_vector_type(8)))  int i32x8;
typedef __attribute__((ext_vector_type(4)))  int i32x4;

#if defined(__has_builtin)
#if __has_builtin(__builtin_amdgcn_tensor_load_to_lds) && \
    __has_builtin(__builtin_amdgcn_s_wait_tensorcnt)
#define HAVE_TDM 1
#endif
#endif
#ifndef HAVE_TDM
#define HAVE_TDM 0
#endif

union FragAB {
  v16bf v;
  uint4 u[2];
};

struct U16x4 { unsigned short x, y, z, w; };

__device__ __forceinline__ unsigned short f2bf(float f) {
  unsigned u = __builtin_bit_cast(unsigned, f);
  u += 0x7FFFu + ((u >> 16) & 1u);   // round-to-nearest-even
  return (unsigned short)(u >> 16);
}

#if HAVE_TDM
// Issue one TDM 2-D tile load (bf16 elements) into LDS.
//   tile_w elems along dim0 (contiguous), tile_h rows, row stride = `stride`
//   elems. D# packing per CDNA5 ISA ch.8 (groups 2/3 zero => 2-D tensor).
// Toolchain builtin arity: 6 args (g0, g1, g2, g3, g4, cpol).
__device__ __forceinline__ void tdm_load_2d(unsigned lds_addr,
                                            const void* gtile,
                                            unsigned tile_w, unsigned tile_h,
                                            unsigned tensor_w,
                                            unsigned tensor_h,
                                            unsigned stride) {
  unsigned long long ga = (unsigned long long)(size_t)gtile;
  u32x4 g0;
  g0.x = 1u;                                        // count=1, user mode
  g0.y = lds_addr;                                  // lds_addr [63:32]
  g0.z = (unsigned)ga;                              // global_addr lo
  g0.w = (unsigned)((ga >> 32) & 0x01FFFFFFu)       // global_addr [56:32]
         | (2u << 30);                              // type = 2 ("image")
  i32x8 g1;
  g1[0] = (int)(1u << 16);                          // data_size=1 (2 bytes)
  g1[1] = (int)((tensor_w & 0xFFFFu) << 16);        // tensor_dim0 [15:0]
  g1[2] = (int)((tensor_w >> 16) | ((tensor_h & 0xFFFFu) << 16));
  g1[3] = (int)((tensor_h >> 16) | (tile_w << 16)); // tile_dim0
  g1[4] = (int)(tile_h & 0xFFFFu);                  // tile_dim1 (tile_dim2=0)
  g1[5] = (int)stride;                              // tensor_dim0_stride lo
  g1[6] = 0;
  g1[7] = 0;
  i32x4 z4 = {0, 0, 0, 0};
  i32x8 z8 = {0, 0, 0, 0, 0, 0, 0, 0};
  __builtin_amdgcn_tensor_load_to_lds(g0, g1, z4, z4, z8, 0);
}
#endif

// ---------------------------------------------------------------------------
// fp32 -> bf16 conversion, 4 elements / thread
// ---------------------------------------------------------------------------
__global__ void cvt_f32_bf16_v4(const float4* __restrict__ src,
                                U16x4* __restrict__ dst, int n4) {
  int i = blockIdx.x * blockDim.x + threadIdx.x;
  if (i < n4) {
    float4 f = src[i];
    U16x4 o;
    o.x = f2bf(f.x); o.y = f2bf(f.y); o.z = f2bf(f.z); o.w = f2bf(f.w);
    dst[i] = o;
  }
}

// ---------------------------------------------------------------------------
// C[M,N] = A[M,K](bf16) * W[N,K]^T(bf16) + bias.  M = 4096 for all calls.
// Block: 256 threads = 8 waves; block tile 64(M) x 128(N); wave tile 16x64.
// A/B tiles (64x32, 128x32) double-buffered in LDS, streamed by the TDM
// (wave 0 issues tensor_load_to_lds; all waves sync on TENSORcnt+barrier),
// fragments read from LDS as 16B ds loads per the 16-bit WMMA striping.
// ---------------------------------------------------------------------------
template <bool OUT_BF16>
__global__ __launch_bounds__(256) void gemm_bf16_wmma(
    const unsigned short* __restrict__ A,
    const unsigned short* __restrict__ W,
    const float* __restrict__ bias,
    void* __restrict__ Cout, int N, int K) {
  __shared__ __align__(16) unsigned short Abuf[2][64 * 32];   //  8 KB
  __shared__ __align__(16) unsigned short Bbuf[2][128 * 32];  // 16 KB

  const int lane = threadIdx.x & 31;
  const int wid  = threadIdx.x >> 5;
  const int l15  = lane & 15;
  const int kb   = ((lane >> 4) & 1) * 8;  // K chunk base inside fragment
  const int mBlk = blockIdx.x * 64;
  const int nBlk = blockIdx.y * 128;
  const int mW   = (wid & 3) * 16;   // wave rows within block tile
  const int nW   = (wid >> 2) * 64;  // wave cols within block tile

  v8f acc[4];
#pragma unroll
  for (int t = 0; t < 4; ++t)
#pragma unroll
    for (int r = 0; r < 8; ++r) acc[t][r] = 0.0f;

#if HAVE_TDM
  if (threadIdx.x < 32) {
    tdm_load_2d((unsigned)(size_t)&Abuf[0][0],
                A + (size_t)mBlk * K, 32, 64, (unsigned)K, 4096u, (unsigned)K);
    tdm_load_2d((unsigned)(size_t)&Bbuf[0][0],
                W + (size_t)nBlk * K, 32, 128, (unsigned)K, (unsigned)N,
                (unsigned)K);
  }
#endif

  for (int k0 = 0; k0 < K; k0 += 32) {
#if HAVE_TDM
    const int cur = (k0 >> 5) & 1;
    if (threadIdx.x < 32) __builtin_amdgcn_s_wait_tensorcnt(0);
    __syncthreads();  // current buffers resident for all waves
    if (k0 + 32 < K && threadIdx.x < 32) {
      const int nxt = cur ^ 1;
      tdm_load_2d((unsigned)(size_t)&Abuf[nxt][0],
                  A + (size_t)mBlk * K + (k0 + 32), 32, 64, (unsigned)K, 4096u,
                  (unsigned)K);
      tdm_load_2d((unsigned)(size_t)&Bbuf[nxt][0],
                  W + (size_t)nBlk * K + (k0 + 32), 32, 128, (unsigned)K,
                  (unsigned)N, (unsigned)K);
    }
#else
    const int cur = 0;
    __syncthreads();  // previous WMMA reads done before overwrite
    {
      const int id = threadIdx.x;
      {  // A tile: 64 rows x 64B = 256 x 16B chunks
        const int row = id >> 2, c = id & 3;
        *(uint4*)&Abuf[0][row * 32 + c * 8] =
            *(const uint4*)(A + (size_t)(mBlk + row) * K + k0 + c * 8);
      }
#pragma unroll
      for (int i = 0; i < 2; ++i) {  // B tile: 128 rows => 512 chunks
        const int e = i * 256 + id;
        const int row = e >> 2, c = e & 3;
        *(uint4*)&Bbuf[0][row * 32 + c * 8] =
            *(const uint4*)(W + (size_t)(nBlk + row) * K + k0 + c * 8);
      }
    }
    __syncthreads();
#endif

    FragAB a;
    const unsigned short* ap = &Abuf[cur][(mW + l15) * 32 + kb];
    a.u[0] = *(const uint4*)(ap);
    a.u[1] = *(const uint4*)(ap + 16);
#pragma unroll
    for (int t = 0; t < 4; ++t) {
      FragAB b;
      const unsigned short* bp = &Bbuf[cur][(nW + t * 16 + l15) * 32 + kb];
      b.u[0] = *(const uint4*)(bp);
      b.u[1] = *(const uint4*)(bp + 16);
      acc[t] = __builtin_amdgcn_wmma_f32_16x16x32_bf16(
          false, a.v, false, b.v, (short)0, acc[t], false, false);
    }
  }

  const int rb = ((lane >> 4) & 1) * 8;  // C/D: row = rb + vgpr index
#pragma unroll
  for (int t = 0; t < 4; ++t) {
    const int col = nBlk + nW + t * 16 + l15;
    const float bv = bias[col];
#pragma unroll
    for (int r = 0; r < 8; ++r) {
      const int row = mBlk + mW + rb + r;
      const float v = acc[t][r] + bv;
      if (OUT_BF16)
        ((unsigned short*)Cout)[(size_t)row * N + col] = f2bf(v);
      else
        ((float*)Cout)[(size_t)row * N + col] = v;
    }
  }
}

// ---------------------------------------------------------------------------
// Attention core: one workgroup per (b, h, 32-row q tile).
// Phase 1: S(32x1024) = Q*K^T * scale, mask fused, kept f32 in LDS (128KB).
// Phase 2: row softmax in LDS, probabilities stored bf16 in LDS (64KB).
// Phase 3: ctx(32x64) = P*V; V staged transposed through LDS each 32-k step.
// ---------------------------------------------------------------------------
#define LQ 32

__global__ __launch_bounds__(256) void attn_kernel(
    const unsigned short* __restrict__ Q,
    const unsigned short* __restrict__ Kmat,
    const unsigned short* __restrict__ Vmat,
    const int* __restrict__ mask,
    unsigned short* __restrict__ Ctx) {
  __shared__ __align__(16) float          S[LQ * 1024];       // 128 KB
  __shared__ __align__(16) unsigned short P[LQ * 1024];       //  64 KB
  __shared__ __align__(16) unsigned short Vt[64 * 40];        //   5 KB
  __shared__ float redbuf[LQ * 8];

  const int qt = blockIdx.x;   // q tile: 0..31
  const int h  = blockIdx.y;   // head:   0..15
  const int b  = blockIdx.z;   // batch:  0..3
  const int q0 = qt * LQ;

  const int lane = threadIdx.x & 31;
  const int wid  = threadIdx.x >> 5;
  const int l15  = lane & 15;
  const int kb   = ((lane >> 4) & 1) * 8;
  const int rb   = kb;  // C-layout row base uses same expression

  // ---------------- Phase 1: masked scaled scores ----------------
  const int m_off = (wid & 1) * 16;    // 2 row halves
  const int n_off = (wid >> 1) * 256;  // 4 col quarters

  FragAB qa[2];
  {
    const unsigned short* qp =
        Q + (size_t)(b * 1024 + q0 + m_off + l15) * 1024 + h * 64;
#pragma unroll
    for (int s = 0; s < 2; ++s) {
      qa[s].u[0] = *(const uint4*)(qp + s * 32 + kb);
      qa[s].u[1] = *(const uint4*)(qp + s * 32 + kb + 16);
    }
  }
  const float scale = 0.03125f;  // 1/sqrt(1024)

  for (int nt = 0; nt < 16; ++nt) {
    const int kcol0 = n_off + nt * 16;
    v8f acc;
#pragma unroll
    for (int r = 0; r < 8; ++r) acc[r] = 0.0f;
#pragma unroll
    for (int s = 0; s < 2; ++s) {
      FragAB bf;
      const unsigned short* kp =
          Kmat + (size_t)(b * 1024 + kcol0 + l15) * 1024 + h * 64 + s * 32 + kb;
      bf.u[0] = *(const uint4*)(kp);
      bf.u[1] = *(const uint4*)(kp + 16);
      acc = __builtin_amdgcn_wmma_f32_16x16x32_bf16(
          false, qa[s].v, false, bf.v, (short)0, acc, false, false);
    }
    const int kcol = kcol0 + l15;
#pragma unroll
    for (int r = 0; r < 8; ++r) {
      const int lr = m_off + rb + r;
      const int gq = q0 + lr;
      const int mv = mask[((size_t)b * 1024 + gq) * 1024 + kcol];
      S[lr * 1024 + kcol] = mv ? acc[r] * scale : -1.0e9f;
    }
  }
  __syncthreads();

  // ---------------- Phase 2: softmax (8 threads / row) ----------------
  {
    const int row = threadIdx.x >> 3;  // 0..31
    const int seg = threadIdx.x & 7;   // 0..7
    float* srow = &S[row * 1024 + seg * 128];

    float lmax = -3.0e38f;
    for (int j = 0; j < 128; ++j) lmax = fmaxf(lmax, srow[j]);
    redbuf[row * 8 + seg] = lmax;
    __syncthreads();
    float rmax = redbuf[row * 8];
#pragma unroll
    for (int s = 1; s < 8; ++s) rmax = fmaxf(rmax, redbuf[row * 8 + s]);
    __syncthreads();

    float lsum = 0.0f;
    for (int j = 0; j < 128; ++j) {
      float e = __expf(srow[j] - rmax);
      srow[j] = e;
      lsum += e;
    }
    redbuf[row * 8 + seg] = lsum;
    __syncthreads();
    float rsum = 0.0f;
#pragma unroll
    for (int s = 0; s < 8; ++s) rsum += redbuf[row * 8 + s];
    const float inv = 1.0f / rsum;

    unsigned short* prow = &P[row * 1024 + seg * 128];
    for (int j = 0; j < 128; ++j) prow[j] = f2bf(srow[j] * inv);
  }
  __syncthreads();

  // ---------------- Phase 3: ctx = P * V ----------------
  const int m2 = (wid & 1) * 16;   // output rows (of 32)
  const int d2 = (wid >> 1) * 16;  // output cols (of 64)
  v8f acc;
#pragma unroll
  for (int r = 0; r < 8; ++r) acc[r] = 0.0f;

  for (int j0 = 0; j0 < 1024; j0 += 32) {
    __syncthreads();
    // stage V^T tile: Vt[d][j'] (row stride 40 elems => 16B-aligned chunks)
#pragma unroll
    for (int i = 0; i < 8; ++i) {
      const int e  = i * 256 + threadIdx.x;
      const int d  = e & 63;
      const int jp = e >> 6;
      Vt[d * 40 + jp] =
          Vmat[(size_t)(b * 1024 + j0 + jp) * 1024 + h * 64 + d];
    }
    __syncthreads();

    FragAB a, bf;
    const unsigned short* pp = &P[(m2 + l15) * 1024 + j0 + kb];
    a.u[0] = *(const uint4*)(pp);
    a.u[1] = *(const uint4*)(pp + 16);
    const unsigned short* vp = &Vt[(d2 + l15) * 40 + kb];
    bf.u[0] = *(const uint4*)(vp);
    bf.u[1] = *(const uint4*)(vp + 16);
    acc = __builtin_amdgcn_wmma_f32_16x16x32_bf16(
        false, a.v, false, bf.v, (short)0, acc, false, false);
  }

#pragma unroll
  for (int r = 0; r < 8; ++r) {
    const int gq = q0 + m2 + rb + r;
    const int d  = d2 + l15;
    Ctx[(size_t)(b * 1024 + gq) * 1024 + h * 64 + d] = f2bf(acc[r]);
  }
}

// ---------------------------------------------------------------------------
// Host launcher.  d_in: x, mask, Wq, bq, Wk, bk, Wv, bv, Wo, bo
// Workspace (bytes): 0 xb(8MB) | 8M Wqb | 10M Wkb | 12M Wvb | 14M Wob |
//                    16M Qb(8MB) | 24M Kb | 32M Vb | 40M Ctxb  (48MB total)
// ---------------------------------------------------------------------------
extern "C" void kernel_launch(void* const* d_in, const int* in_sizes, int n_in,
                              void* d_out, int out_size, void* d_ws,
                              size_t ws_size, hipStream_t stream) {
  const float* x  = (const float*)d_in[0];
  const int*   mk = (const int*)d_in[1];
  const float* Wq = (const float*)d_in[2];
  const float* bq = (const float*)d_in[3];
  const float* Wk = (const float*)d_in[4];
  const float* bk = (const float*)d_in[5];
  const float* Wv = (const float*)d_in[6];
  const float* bv = (const float*)d_in[7];
  const float* Wo = (const float*)d_in[8];
  const float* bo = (const float*)d_in[9];
  float* out = (float*)d_out;

  char* ws = (char*)d_ws;
  const size_t MB = 1024ull * 1024ull;
  unsigned short* xb   = (unsigned short*)(ws + 0);
  unsigned short* Wqb  = (unsigned short*)(ws + 8 * MB);
  unsigned short* Wkb  = (unsigned short*)(ws + 10 * MB);
  unsigned short* Wvb  = (unsigned short*)(ws + 12 * MB);
  unsigned short* Wob  = (unsigned short*)(ws + 14 * MB);
  unsigned short* Qb   = (unsigned short*)(ws + 16 * MB);
  unsigned short* Kb   = (unsigned short*)(ws + 24 * MB);
  unsigned short* Vb   = (unsigned short*)(ws + 32 * MB);
  unsigned short* Ctxb = (unsigned short*)(ws + 40 * MB);

  // 1) fp32 -> bf16 conversions
  cvt_f32_bf16_v4<<<4096, 256, 0, stream>>>((const float4*)x, (U16x4*)xb,
                                            4096 * 1024 / 4);
  cvt_f32_bf16_v4<<<1024, 256, 0, stream>>>((const float4*)Wq, (U16x4*)Wqb,
                                            1024 * 1024 / 4);
  cvt_f32_bf16_v4<<<1024, 256, 0, stream>>>((const float4*)Wk, (U16x4*)Wkb,
                                            1024 * 1024 / 4);
  cvt_f32_bf16_v4<<<1024, 256, 0, stream>>>((const float4*)Wv, (U16x4*)Wvb,
                                            1024 * 1024 / 4);
  cvt_f32_bf16_v4<<<1024, 256, 0, stream>>>((const float4*)Wo, (U16x4*)Wob,
                                            1024 * 1024 / 4);

  // 2) Q/K/V projections (bf16 out)
  dim3 ggrid(4096 / 64, 1024 / 128);
  gemm_bf16_wmma<true><<<ggrid, 256, 0, stream>>>(xb, Wqb, bq, Qb, 1024, 1024);
  gemm_bf16_wmma<true><<<ggrid, 256, 0, stream>>>(xb, Wkb, bk, Kb, 1024, 1024);
  gemm_bf16_wmma<true><<<ggrid, 256, 0, stream>>>(xb, Wvb, bv, Vb, 1024, 1024);

  // 3) attention core: grid (q tiles, heads, batch)
  attn_kernel<<<dim3(1024 / LQ, 16, 4), 256, 0, stream>>>(Qb, Kb, Vb, mk,
                                                          Ctxb);

  // 4) output projection (f32 out)
  gemm_bf16_wmma<false><<<ggrid, 256, 0, stream>>>(Ctxb, Wob, bo, out, 1024,
                                                   1024);
}
